// ClusterisedSelfAttentionNotLearnable_47107201303344
// MI455X (gfx1250) — matrix operand order repaired and checked by reading, not compiled, not run on hardware
//
#include <hip/hip_runtime.h>

typedef _Float16 v16h __attribute__((ext_vector_type(16)));
typedef _Float16 v8h  __attribute__((ext_vector_type(8)));
typedef _Float16 v4h  __attribute__((ext_vector_type(4)));
typedef _Float16 v2h  __attribute__((ext_vector_type(2)));
typedef float    v8f  __attribute__((ext_vector_type(8)));
typedef float    v4f  __attribute__((ext_vector_type(4)));

#define NFREQ 6
#define E_REAL 72
#define EPAD 104          // padded enc length; 52-dword LDS row stride (conflict-free b128 loads)
#define NCOL 768          // 3*C
#define NCLUS 256
#define WAVES 8
#define ROWS_PER_WAVE 32  // two 16-row WMMA tiles per wave
#define ROWS_PER_BLOCK (WAVES * ROWS_PER_WAVE)   // 256

// dynamic-LDS carve-up (bytes)
#define OFF_WH    0
#define SZ_WH     (NCOL * EPAD * 2)                    // 159744: W as f16, padded rows
#define OFF_CEN   (OFF_WH + SZ_WH)
#define SZ_CEN    (NCLUS * 3 * 4)                      // 3072
#define OFF_ATTN  (OFF_CEN + SZ_CEN)
#define SZ_ATTN   (WAVES * ROWS_PER_WAVE * NCLUS * 2)  // 131072: softmax weights as f16
#define OFF_RSUM  (OFF_ATTN + SZ_ATTN)
#define SZ_RSUM   (WAVES * ROWS_PER_WAVE * 4)          // 1024
#define SMEM_BYTES (OFF_RSUM + SZ_RSUM)                // 294912 <= 320KB WGP LDS

__device__ __forceinline__ float encval(const float* xv, int k) {
  // posenc: per dim d (=k/12): [sin(2^0 x)..sin(2^5 x), cos(2^0 x)..cos(2^5 x)]
  if (k >= E_REAL) return 0.0f;
  const int d = k / 12;
  const int j = k % 12;
  const float ang = xv[d] * (float)(1 << (j % NFREQ));
  return (j < NFREQ) ? __sinf(ang) : __cosf(ang);
}

// CDNA5 16-bit 16x32 A layout: lanes 0-15 (KOFF=0): halves 0-7 -> K=kc*32+0..7,
// halves 8-15 -> K=kc*32+16..23; lanes 16-31 (KOFF=8): shifted by +8.
template <int KOFF>
__device__ __forceinline__ void build_a(const float* xv, v16h a[3]) {
#pragma unroll
  for (int kc = 0; kc < 3; ++kc) {
#pragma unroll
    for (int h = 0; h < 16; ++h) {
      const int k = kc * 32 + (h < 8 ? h : 8 + h) + KOFF;
      a[kc][h] = (_Float16)encval(xv, k);
    }
  }
}

__global__ void __launch_bounds__(256, 1)
fused_posenc_wmma_attn(const float* __restrict__ X,
                       const float* __restrict__ W,     // [768, 72] row-major
                       const float* __restrict__ cen,   // [256, 3]
                       float* __restrict__ out,         // [N, 3]
                       int N) {
  extern __shared__ char smem[];
  _Float16* Wh    = (_Float16*)(smem + OFF_WH);
  float*    cenS  = (float*)(smem + OFF_CEN);
  _Float16* attnH = (_Float16*)(smem + OFF_ATTN);
  float*    rsumS = (float*)(smem + OFF_RSUM);

  const int tid  = threadIdx.x;
  const int wv   = tid >> 5;
  const int lane = tid & 31;
  const int lnid = lane & 15;
  const int half = lane >> 4;

  // ---- stage W: f32 global -> f16 LDS, vectorized ----
  for (int r = tid; r < NCOL; r += 256) {
    const v4f* src = (const v4f*)(W + r * E_REAL);   // 288B rows, 16B aligned
    v4h* dst = (v4h*)(Wh + r * EPAD);                // 208B rows, 8B aligned
#pragma unroll
    for (int q = 0; q < E_REAL / 4; ++q) {
      const v4f f = src[q];
      v4h h;
      h[0] = (_Float16)f[0]; h[1] = (_Float16)f[1];
      h[2] = (_Float16)f[2]; h[3] = (_Float16)f[3];
      dst[q] = h;
    }
    const v4h z = {};
#pragma unroll
    for (int q = E_REAL / 4; q < EPAD / 4; ++q) dst[q] = z;
  }
  for (int i = tid; i < NCLUS * 3; i += 256) cenS[i] = cen[i];
  __syncthreads();

  // ---- per-wave 32-row tile pair ----
  const int r0 = blockIdx.x * ROWS_PER_BLOCK + wv * ROWS_PER_WAVE;
  int rA = r0 + lnid;      if (rA > N - 1) rA = N - 1;   // tile 0 row
  int rB = r0 + 16 + lnid; if (rB > N - 1) rB = N - 1;   // tile 1 row
  float xv0[6], xv1[6];
  {
    const float* pa = X + rA * 6;
    const float* pb = X + rB * 6;
#pragma unroll
    for (int i = 0; i < 6; ++i) { xv0[i] = pa[i]; xv1[i] = pb[i]; }
  }

  // ---- positional encoding straight into WMMA A fragments (K=96, 3 steps) ----
  v16h a0[3], a1[3];
  if (half == 0) { build_a<0>(xv0, a0); build_a<0>(xv1, a1); }
  else           { build_a<8>(xv0, a0); build_a<8>(xv1, a1); }

  // ---- softmax over 256 clusters, one lane per row (lane's row = r0 + lane) ----
  {
    const float xs0 = half ? xv1[0] : xv0[0];
    const float xs1 = half ? xv1[1] : xv0[1];
    const float xs2 = half ? xv1[2] : xv0[2];
    float mx = -3.0e38f;
    for (int c = 0; c < NCLUS; ++c) {
      const float* ce = cenS + c * 3;
      const float s = fmaf(xs0, ce[0], fmaf(xs1, ce[1], xs2 * ce[2]));
      mx = fmaxf(mx, s);
    }
    float sum = 0.0f;
    _Float16* arow = attnH + (wv * ROWS_PER_WAVE + lane) * NCLUS;
    for (int c = 0; c < NCLUS; c += 2) {
      const float* ce = cenS + c * 3;
      const float s0 = fmaf(xs0, ce[0], fmaf(xs1, ce[1], xs2 * ce[2]));
      const float s1 = fmaf(xs0, ce[3], fmaf(xs1, ce[4], xs2 * ce[5]));
      const float e0 = __expf(s0 - mx);
      const float e1 = __expf(s1 - mx);
      v2h pk; pk[0] = (_Float16)e0; pk[1] = (_Float16)e1;
      *(v2h*)(arow + c) = pk;                 // packed b32 store
      sum += e0 + e1;
    }
    rsumS[wv * ROWS_PER_WAVE + lane] = sum;   // unnormalized; divide at the end
  }

  // ---- fused GEMM (enc @ W^T) + attention contraction, software pipelined ----
  // Channel of column n: (n % 3) = (chunk + lnid) % 3. With chunk unrolled by 6,
  // slot u%3 is compile-time: one FMA per (row, chunk); rotation undone once at end.
  float part0[8][3], part1[8][3];
#pragma unroll
  for (int j = 0; j < 8; ++j)
#pragma unroll
    for (int s = 0; s < 3; ++s) { part0[j][s] = 0.0f; part1[j][s] = 0.0f; }

  const _Float16* attA = attnH + (wv * ROWS_PER_WAVE + half * 8) * NCLUS;       // tile 0
  const _Float16* attB = attnH + (wv * ROWS_PER_WAVE + 16 + half * 8) * NCLUS;  // tile 1

  v8h b[2][6];                               // ping-pong B fragments
  {
    const _Float16* wrow = Wh + lnid * EPAD + half * 8;    // chunk 0
#pragma unroll
    for (int kc = 0; kc < 3; ++kc) {
      b[0][2 * kc]     = *(const v8h*)(wrow + kc * 32);
      b[0][2 * kc + 1] = *(const v8h*)(wrow + kc * 32 + 16);
    }
  }

  for (int it = 0; it < 8; ++it) {           // 8 x 6 = 48 column tiles
#pragma unroll
    for (int u = 0; u < 6; ++u) {
      const int chunk = it * 6 + u;
      const int cur = u & 1, nxt = cur ^ 1;

      // prefetch next chunk's B (chunk-48 over-read stays inside the LDS alloc)
      {
        const _Float16* wrow = Wh + ((chunk + 1) * 16 + lnid) * EPAD + half * 8;
#pragma unroll
        for (int kc = 0; kc < 3; ++kc) {
          b[nxt][2 * kc]     = *(const v8h*)(wrow + kc * 32);
          b[nxt][2 * kc + 1] = *(const v8h*)(wrow + kc * 32 + 16);
        }
      }

      const int n = chunk * 16 + lnid;       // global column == W row
      const int cIdx = n / 3;

      float aw0[8], aw1[8];                  // f16 attention weights -> f32
#pragma unroll
      for (int j = 0; j < 8; ++j) {
        aw0[j] = (float)attA[j * NCLUS + cIdx];
        aw1[j] = (float)attB[j * NCLUS + cIdx];
      }

      v8f acc0 = {}, acc1 = {};
#pragma unroll
      for (int kc = 0; kc < 3; ++kc) {
        v16h bb;
#pragma unroll
        for (int t = 0; t < 8; ++t) { bb[t] = b[cur][2 * kc][t]; bb[t + 8] = b[cur][2 * kc + 1][t]; }
        acc0 = __builtin_amdgcn_wmma_f32_16x16x32_f16(false, a0[kc], false, bb, (short)0, acc0, false, false);
        acc1 = __builtin_amdgcn_wmma_f32_16x16x32_f16(false, a1[kc], false, bb, (short)0, acc1, false, false);
      }

      const int slot = u % 3;                // compile-time rotated channel slot
#pragma unroll
      for (int j = 0; j < 8; ++j) {
        part0[j][slot] = fmaf(aw0[j], acc0[j], part0[j][slot]);
        part1[j][slot] = fmaf(aw1[j], acc1[j], part1[j][slot]);
      }

#if __has_builtin(__builtin_amdgcn_sched_group_barrier)
      // pipeline spec per body: 22 DS-reads -> 6 WMMA -> 32 VALU (cvt + fmac)
      __builtin_amdgcn_sched_group_barrier(0x100, 22, 0);  // DS read
      __builtin_amdgcn_sched_group_barrier(0x008, 6, 0);   // MFMA/WMMA
      __builtin_amdgcn_sched_group_barrier(0x002, 32, 0);  // VALU
#endif
    }
  }

  // ---- undo per-lane channel rotation: channel d lives in slot (d - p) mod 3 ----
  const int p = lnid % 3;
  float ch0[8][3], ch1[8][3];
#pragma unroll
  for (int j = 0; j < 8; ++j) {
#pragma unroll
    for (int d = 0; d < 3; ++d) {
      int u = d - p; if (u < 0) u += 3;
      ch0[j][d] = (u == 0) ? part0[j][0] : ((u == 1) ? part0[j][1] : part0[j][2]);
      ch1[j][d] = (u == 0) ? part1[j][0] : ((u == 1) ? part1[j][1] : part1[j][2]);
    }
  }

  // ---- butterfly-reduce across the 16 lanes of each half, then store rgb ----
#pragma unroll
  for (int j = 0; j < 8; ++j)
#pragma unroll
    for (int d = 0; d < 3; ++d) {
      float v = ch0[j][d];
      v += __shfl_xor(v, 1, 32); v += __shfl_xor(v, 2, 32);
      v += __shfl_xor(v, 4, 32); v += __shfl_xor(v, 8, 32);
      ch0[j][d] = v;
      float w = ch1[j][d];
      w += __shfl_xor(w, 1, 32); w += __shfl_xor(w, 2, 32);
      w += __shfl_xor(w, 4, 32); w += __shfl_xor(w, 8, 32);
      ch1[j][d] = w;
    }

#pragma unroll
  for (int j = 0; j < 8; ++j) {
    if (lnid == j) {
      const int m = j + half * 8;
      const int row0 = r0 + m;
      if (row0 < N) {
        const float inv = 1.0f / rsumS[wv * ROWS_PER_WAVE + m];
        out[row0 * 3 + 0] = ch0[j][0] * inv;
        out[row0 * 3 + 1] = ch0[j][1] * inv;
        out[row0 * 3 + 2] = ch0[j][2] * inv;
      }
      const int row1 = r0 + 16 + m;
      if (row1 < N) {
        const float inv = 1.0f / rsumS[wv * ROWS_PER_WAVE + 16 + m];
        out[row1 * 3 + 0] = ch1[j][0] * inv;
        out[row1 * 3 + 1] = ch1[j][1] * inv;
        out[row1 * 3 + 2] = ch1[j][2] * inv;
      }
    }
  }
}

extern "C" void kernel_launch(void* const* d_in, const int* in_sizes, int n_in,
                              void* d_out, int out_size, void* d_ws, size_t ws_size,
                              hipStream_t stream) {
  (void)d_ws; (void)ws_size; (void)out_size;
  if (n_in < 3) return;
  const float* X   = (const float*)d_in[0];   // [N, 6]
  const float* W   = (const float*)d_in[1];   // [768, 72]
  const float* cen = (const float*)d_in[2];   // [256, 3]
  float* out = (float*)d_out;                 // [N, 3]

  const int N = in_sizes[0] / 6;
  const int grid = (N + ROWS_PER_BLOCK - 1) / ROWS_PER_BLOCK;

  hipFuncSetAttribute((const void*)fused_posenc_wmma_attn,
                      hipFuncAttributeMaxDynamicSharedMemorySize,
                      (int)SMEM_BYTES);
  hipLaunchKernelGGL(fused_posenc_wmma_attn, dim3(grid), dim3(256),
                     SMEM_BYTES, stream, X, W, cen, out, N);
}
